// RNN_10453950398947
// MI455X (gfx1250) — matrix operand order repaired and checked by reading, not compiled, not run on hardware
//
#include <hip/hip_runtime.h>
#include <math.h>

// Problem constants (match reference)
#define B_  64
#define T_  2048
#define I_  128
#define H_  512
#define O_  64
#define KC  (H_ + I_)      // 640 : fused K = [r | x]
#define KCP (KC + 8)       // 648 : padded LDS row (648*2/4 = 324 words, 324 % 64 = 4 -> staggered banks)
#define HP  (H_ + 4)       // 516 : padded noise row (516 % 64 = 4; 8-row gap = 32 banks between lane halves)
#define NOISE_STD 0.0005f
#define ALPHA     0.1f

typedef __bf16 bf16;
typedef __attribute__((ext_vector_type(16))) bf16         v16bf;
typedef __attribute__((ext_vector_type(8)))  float        v8f;
typedef __attribute__((ext_vector_type(4)))  float        v4f;
typedef __attribute__((ext_vector_type(4)))  unsigned int v4u;

static __device__ inline bf16 f2bf(float f) {
  union { float f; unsigned u; } v; v.f = f;
  unsigned r = v.u + 0x7FFFu + ((v.u >> 16) & 1u);   // round-to-nearest-even
  unsigned short h = (unsigned short)(r >> 16);
  return __builtin_bit_cast(bf16, h);
}

// CDNA5 has a hardware TANH transcendental (TRANS32 class). Use it if the
// builtin is exposed; else a 2-TRANS formulation (v_exp_f32 + v_rcp_f32),
// both far beyond bf16 accuracy.
static __device__ inline float fast_tanh(float x) {
#if __has_builtin(__builtin_amdgcn_tanhf)
  return __builtin_amdgcn_tanhf(x);
#else
  float a = __builtin_fabsf(x);
  float e = __builtin_amdgcn_exp2f(2.8853900817779268f * a);  // exp(2a)
  float t = 1.f - 2.f * __builtin_amdgcn_rcpf(e + 1.f);
  return __builtin_copysignf(t, x);
#endif
}

// A-fragment: 16x32 bf16, lane = 16*lhi + lmod holds row M=lmod,
// K segments [16*lhi, +8) and [16*lhi+8, +8)  -> two ds_load_b128
static __device__ inline v16bf load_afrag(const bf16* p0, const bf16* p1) {
  union { v16bf v; struct { v4u lo, hi; } w; } u;
  u.w.lo = *(const v4u*)p0;
  u.w.hi = *(const v4u*)p1;
  return u.v;
}

// ---------------------------------------------------------------------------
// Prep: one-time weight conversion + relayout into L2-resident bf16 buffers.
//   wcat [H_][KC]  : row n = [ wrec[n][0..H) | wi[0..I)[n] ]   (B-operand rows)
//   woutT[O_][H_]  : row o = wout[:, o]                        (B-operand rows)
// ---------------------------------------------------------------------------
__global__ void rnn_prep(const float* __restrict__ wi, const float* __restrict__ wrec,
                         const float* __restrict__ wout,
                         bf16* __restrict__ wcat, bf16* __restrict__ woutT) {
  const int total = H_ * KC + O_ * H_;
  for (int idx = blockIdx.x * blockDim.x + threadIdx.x; idx < total;
       idx += gridDim.x * blockDim.x) {
    if (idx < H_ * KC) {
      int n = idx / KC, k = idx - n * KC;
      float v = (k < H_) ? wrec[n * H_ + k] : wi[(k - H_) * H_ + n];
      wcat[idx] = f2bf(v);
    } else {
      int j = idx - H_ * KC;
      int o = j / H_, hh = j - o * H_;
      woutT[j] = f2bf(wout[hh * O_ + o]);
    }
  }
}

// ---------------------------------------------------------------------------
// Persistent RNN scan. 4 workgroups (one 16-row batch tile each), 16 waves.
// h lives in WMMA f32 accumulators for the whole scan. Per step per wave:
// 20 chained v_wmma_f32_16x16x32_bf16 (fused K = [r|x] = 640); waves 0..3
// additionally run the 16-WMMA output GEMM concurrently (same r_t operand).
// 2 barriers per step; all global streams use running pointers; tanh is a
// single TRANS32 hardware op.
// ---------------------------------------------------------------------------
__launch_bounds__(512, 1)
__global__ void rnn_main(const float* __restrict__ input, const float* __restrict__ noise,
                         const float* __restrict__ h0p,
                         const bf16* __restrict__ wcat, const bf16* __restrict__ woutT,
                         float* __restrict__ out) {
  __shared__ bf16  rcat[16 * KCP];   // [b_local][ r(0..511) | x(512..639) ]  ~20.7 KB
  __shared__ float nzs[16 * HP];     // [b_local][h]  staged noise            ~33.0 KB

  const int tid  = threadIdx.x;
  const int lane = tid & 31;
  const int wave = tid >> 5;        // 0..15
  const int lhi  = lane >> 4;       // 0 or 1
  const int lmod = lane & 15;
  const int b0   = blockIdx.x * 16; // batch tile base

  const int n0 = wave * 32 + lmod;  // this lane's two hidden columns
  const int n1 = n0 + 16;
  const bf16* wrow0 = wcat + (size_t)n0 * KC;
  const bf16* wrow1 = wcat + (size_t)n1 * KC;

  // ---- running global pointers (advance by a constant each step) ----------
  // x staging: thread handles (bb = tid>>7 + 4j, i = tid&127), j = 0..3
  const float* xq[4];
  bf16* xl[4];
#pragma unroll
  for (int j = 0; j < 4; ++j) {
    int bb = (tid >> 7) + 4 * j;
    xq[j] = input + (size_t)(b0 + bb) * T_ * I_ + (tid & (I_ - 1));
    xl[j] = &rcat[bb * KCP + H_ + (tid & (I_ - 1))];
  }
  // noise staging: wave w stages batch row w; lane covers 4 x v4f chunks
  const float* np = noise + (size_t)(b0 + wave) * T_ * H_ + lane * 4;
  float* nzrow = &nzs[wave * HP + lane * 4];
  // output: waves 0..3 own O-tile `wave`; per-r offsets fit 24-bit immediates
  float* op = out + (size_t)(b0 + 8 * lhi) * T_ * O_ + wave * 16 + lmod;

  // hidden state resident in accumulator registers: (b_local = r + 8*lhi, n)
  v8f hA, hB;
  {
    float a = h0p[n0], b = h0p[n1];
#pragma unroll
    for (int r = 0; r < 8; ++r) { hA[r] = a; hB[r] = b; }
  }

  auto write_r = [&]() {   // r = tanh(h) -> LDS (each wave writes its slice)
#pragma unroll
    for (int r = 0; r < 8; ++r) {
      int bl = r + 8 * lhi;
      rcat[bl * KCP + n0] = f2bf(fast_tanh(hA[r]));
      rcat[bl * KCP + n1] = f2bf(fast_tanh(hB[r]));
    }
  };

  auto out_step = [&]() {  // out GEMM from current rcat r-part; advances op
    if (wave < 4) {
      const bf16* orow = woutT + (size_t)(wave * 16 + lmod) * H_;
      v8f c = {};
#pragma unroll 4
      for (int kt = 0; kt < H_ / 32; ++kt) {
        int kb = kt * 32;
        v16bf a = load_afrag(&rcat[lmod * KCP + kb + 16 * lhi],
                             &rcat[lmod * KCP + kb + 16 * lhi + 8]);
        v16bf bm = *(const v16bf*)(orow + kb + 16 * lhi);
        c = __builtin_amdgcn_wmma_f32_16x16x32_bf16(false, a, false, bm,
                                                    (short)0, c, false, false);
      }
#pragma unroll
      for (int r = 0; r < 8; ++r)
        op[(size_t)r * T_ * O_] = c[r];   // imm offsets: r*T*O*4 <= 3.67 MB
      op += O_;
    }
  };

  write_r();                               // r_0 = tanh(h0)

  for (int step = 0; step < T_ - 1; ++step) {
    // ---- stage x_t (bf16 -> rcat x-part) and noise_t (f32 -> nzs) --------
#pragma unroll
    for (int j = 0; j < 4; ++j) { *xl[j] = f2bf(*xq[j]); xq[j] += I_; }
#pragma unroll
    for (int c = 0; c < 4; ++c)
      *(v4f*)(nzrow + 128 * c) = *(const v4f*)(np + 128 * c);
    __builtin_prefetch(np + H_, 0, 0);     // next step's noise row
    np += H_;
    __syncthreads();          // staging done; r_t (prev write_r) visible

    // ---- out_t GEMM (waves 0..3) overlapped with the main recurrence -----
    out_step();

    // ---- fused recurrent+input matmul: acc = [r|x](16x640) x Wcat slice --
    v8f acc0 = {}, acc1 = {};
#pragma unroll 4
    for (int kt = 0; kt < KC / 32; ++kt) {
      int kb = kt * 32;
      v16bf a = load_afrag(&rcat[lmod * KCP + kb + 16 * lhi],
                           &rcat[lmod * KCP + kb + 16 * lhi + 8]);
      v16bf bm0 = *(const v16bf*)(wrow0 + kb + 16 * lhi);
      v16bf bm1 = *(const v16bf*)(wrow1 + kb + 16 * lhi);
      acc0 = __builtin_amdgcn_wmma_f32_16x16x32_bf16(false, a, false, bm0,
                                                     (short)0, acc0, false, false);
      acc1 = __builtin_amdgcn_wmma_f32_16x16x32_bf16(false, a, false, bm1,
                                                     (short)0, acc1, false, false);
    }

    // ---- fp32 state update: h = (1-a)h + s*noise + a*acc (noise from LDS)
#pragma unroll
    for (int r = 0; r < 8; ++r) {
      const float* nrow = &nzs[(r + 8 * lhi) * HP];
      hA[r] = (1.f - ALPHA) * hA[r] + NOISE_STD * nrow[n0] + ALPHA * acc0[r];
      hB[r] = (1.f - ALPHA) * hB[r] + NOISE_STD * nrow[n1] + ALPHA * acc1[r];
    }

    __syncthreads();          // all rcat/nzs reads done before overwrite
    write_r();                // r_{t+1}
  }

  __syncthreads();            // r_{T-1} visible
  out_step();                 // out at t = T-1
}

extern "C" void kernel_launch(void* const* d_in, const int* in_sizes, int n_in,
                              void* d_out, int out_size, void* d_ws, size_t ws_size,
                              hipStream_t stream) {
  (void)in_sizes; (void)n_in; (void)out_size; (void)ws_size;
  const float* input = (const float*)d_in[0];
  const float* noise = (const float*)d_in[1];
  const float* wi    = (const float*)d_in[2];
  const float* wrec  = (const float*)d_in[3];
  const float* wout  = (const float*)d_in[4];
  const float* h0    = (const float*)d_in[5];

  bf16* wcat  = (bf16*)d_ws;                       // 512*640*2 = 640 KB
  bf16* woutT = wcat + (size_t)H_ * KC;            // +  64 KB

  rnn_prep<<<1408, 256, 0, stream>>>(wi, wrec, wout, wcat, woutT);
  rnn_main<<<B_ / 16, 512, 0, stream>>>(input, noise, h0, wcat, woutT, (float*)d_out);
}
